// RegularizedSSM_60155311948505
// MI455X (gfx1250) — compile-verified, matrix operand order checked by reading, not run on hardware
//
#include <hip/hip_runtime.h>
#include <hip/hip_bf16.h>

// ---------------------------------------------------------------------------
// RegularizedSSM for MI455X (gfx1250, wave32, WMMA)
//   ys, hs = f(x, W_A, W_B, W_C)
// d_out = [ ys (32*1024*1024 f32) | hs (32*1024*1024 f32) ]
// ---------------------------------------------------------------------------

typedef __attribute__((ext_vector_type(16))) __bf16 v16bf;
typedef __attribute__((ext_vector_type(8)))  float  v8f;
typedef __attribute__((ext_vector_type(2)))  float  v2f;
typedef int v4i_ __attribute__((vector_size(16)));   // matches builtin proto

#define B_DIM 32
#define T_DIM 1024
#define H_DIM 1024
#define GM    (B_DIM * T_DIM)   // 32768 rows for the big GEMMs
#define GK    H_DIM
#define GN    H_DIM

// ---- CDNA5 async global->LDS staging (ASYNCcnt path), with fallback -------
#if defined(__has_builtin)
#if __has_builtin(__builtin_amdgcn_global_load_async_to_lds_b128)
#define HAVE_ASYNC_LDS 1
#endif
#endif

__device__ inline void wait_async0() {
#if defined(__has_builtin) && __has_builtin(__builtin_amdgcn_s_wait_asynccnt)
  __builtin_amdgcn_s_wait_asynccnt(0);
#elif defined(HAVE_ASYNC_LDS)
  asm volatile("s_wait_asynccnt 0" ::: "memory");
#endif
}

__device__ inline void async_copy_b128(const void* g, void* l) {
#if defined(HAVE_ASYNC_LDS)
  __builtin_amdgcn_global_load_async_to_lds_b128(
      (__attribute__((address_space(1))) v4i_*)g,
      (__attribute__((address_space(3))) v4i_*)l, 0, 0);
#else
  *(uint4*)l = *(const uint4*)g;
#endif
}

// ======================= block reduction helper ============================
__device__ inline float block_reduce_sum(float v, float* sred, int tid) {
#pragma unroll
  for (int off = 16; off > 0; off >>= 1) v += __shfl_xor(v, off, 32);
  const int wid = tid >> 5, lane = tid & 31;
  if (lane == 0) sred[wid] = v;
  __syncthreads();
  if (tid < 32) {
    float r = sred[tid];
#pragma unroll
    for (int off = 16; off > 0; off >>= 1) r += __shfl_xor(r, off, 32);
    if (tid == 0) sred[0] = r;
  }
  __syncthreads();
  float out = sred[0];
  __syncthreads();   // sred reused by caller
  return out;
}

// ======================= 1) spectral-norm power iteration ==================
__global__ __launch_bounds__(1024)
void power_iter_kernel(const float* __restrict__ WA,
                       const float* __restrict__ WB,
                       const float* __restrict__ WC,
                       float* __restrict__ inv_sigma) {
  const int n = H_DIM;
  const float* W = (blockIdx.x == 0) ? WA : (blockIdx.x == 1) ? WB : WC;
  __shared__ float su[H_DIM];
  __shared__ float sv[H_DIM];
  __shared__ float sred[32];
  const int tid = threadIdx.x;

  su[tid] = rsqrtf((float)n);
  __syncthreads();

  for (int iter = 0; iter < 10; ++iter) {
    float acc = 0.f;                      // v = normalize(W^T u)
    for (int i = 0; i < n; ++i) acc += W[(size_t)i * n + tid] * su[i];
    float s = block_reduce_sum(acc * acc, sred, tid);
    sv[tid] = acc * (1.0f / (sqrtf(s) + 1e-12f));
    __syncthreads();
    const float* row = W + (size_t)tid * n;  // u = normalize(W v)
    float acc2 = 0.f;
    for (int j = 0; j < n; ++j) acc2 += row[j] * sv[j];
    float s2 = block_reduce_sum(acc2 * acc2, sred, tid);
    su[tid] = acc2 * (1.0f / (sqrtf(s2) + 1e-12f));
    __syncthreads();
  }
  float acc = 0.f;                        // v = normalize(W^T u)
  for (int i = 0; i < n; ++i) acc += W[(size_t)i * n + tid] * su[i];
  float s = block_reduce_sum(acc * acc, sred, tid);
  sv[tid] = acc * (1.0f / (sqrtf(s) + 1e-12f));
  __syncthreads();
  const float* row = W + (size_t)tid * n; // sigma = u . (W v)
  float wv = 0.f;
  for (int j = 0; j < n; ++j) wv += row[j] * sv[j];
  float sigma = block_reduce_sum(su[tid] * wv, sred, tid);
  if (tid == 0) inv_sigma[blockIdx.x] = 1.0f / sigma;
}

// ======================= 2) scale + pack weights ===========================
__global__ void pack_kernel(const float* __restrict__ WA,
                            const float* __restrict__ WB,
                            const float* __restrict__ WC,
                            const float* __restrict__ inv_sigma,
                            float* __restrict__ An,
                            __bf16* __restrict__ Bb,
                            __bf16* __restrict__ Cb,
                            unsigned* __restrict__ sync) {
  const int i = blockIdx.x * blockDim.x + threadIdx.x;
  if (i < T_DIM) sync[i] = 0u;           // reset scan-step counters
  if (i < H_DIM * H_DIM) {
    const float ia = inv_sigma[0], ib = inv_sigma[1], ic = inv_sigma[2];
    An[i] = WA[i] * ia;
    Bb[i] = (__bf16)(WB[i] * ib);
    Cb[i] = (__bf16)(WC[i] * ic);
  }
}

// ======================= bf16 WMMA fragment load from LDS ==================
// 16-bit A/B 16x32 layout (ISA 7.12.2): lane<16 -> row=lane, K={0..7,16..23};
// lane>=16 -> row=lane-16, K={8..15,24..31}. Two ds_load_b128 per fragment.
__device__ inline v16bf load_frag_bf16(const __bf16* p, int row0) {
  const int lane = threadIdx.x & 31;
  const int r  = row0 + (lane & 15);
  const int kk = (lane >> 4) << 3;               // 0 or 8
  const __bf16* q = p + r * 32 + kk;
  union { v16bf v; uint4 u[2]; } f;
  f.u[0] = *(const uint4*)q;                     // K kk..kk+7
  f.u[1] = *(const uint4*)(q + 16);              // K kk+16..kk+23
  return f.v;
}

// ======================= 3) big GEMM: out = A(f32) * W(bf16,[N,K])^T =======
// 128x128x32 block tile, double-buffered LDS, one barrier per K-step.
// W tile: async global->LDS (ASYNCcnt). A tile: load + v_cvt_pk_bf16 + ds.
// 512 threads = 16 waves (4x4); each wave: 2x2 v_wmma_f32_16x16x32_bf16.
__global__ __launch_bounds__(512)
void gemm_xW_bf16(const float* __restrict__ A,
                  const __bf16* __restrict__ W,
                  float* __restrict__ out) {
  __shared__ __align__(16) __bf16 sA[2][128 * 32];
  __shared__ __align__(16) __bf16 sB[2][128 * 32];
  const int tid   = threadIdx.x;
  const int mBase = blockIdx.x * 128;
  const int nBase = blockIdx.y * 128;
  const int wid   = tid >> 5;
  const int waveM = wid >> 2, waveN = wid & 3;
  const int srow  = tid >> 2;                    // 0..127
  const int scol  = (tid & 3) << 3;              // 0,8,16,24
  const float*  gA = A + (size_t)(mBase + srow) * GK + scol;
  const __bf16* gW = W + (size_t)(nBase + srow) * GK + scol;

  auto stage = [&](int buf, int k0) {
    // weight tile: straight async DMA into LDS (already bf16)
    async_copy_b128(gW + k0, &sB[buf][srow * 32 + scol]);
    // activation tile: f32 -> bf16 in-register, then LDS
    float4 f0 = *(const float4*)(gA + k0);
    float4 f1 = *(const float4*)(gA + k0 + 4);
    union { __bf16 h[8]; uint4 q; } t;
    t.h[0] = (__bf16)f0.x; t.h[1] = (__bf16)f0.y;
    t.h[2] = (__bf16)f0.z; t.h[3] = (__bf16)f0.w;
    t.h[4] = (__bf16)f1.x; t.h[5] = (__bf16)f1.y;
    t.h[6] = (__bf16)f1.z; t.h[7] = (__bf16)f1.w;
    *(uint4*)&sA[buf][srow * 32 + scol] = t.q;
  };

  stage(0, 0);
  v8f acc[2][2] = {};
  for (int k0 = 0, it = 0; k0 < GK; k0 += 32, ++it) {
    const int cur = it & 1;
    wait_async0();          // this wave's async stores to LDS landed
    __syncthreads();        // everyone's staging of `cur` visible
    if (k0 + 32 < GK) {
      stage(cur ^ 1, k0 + 32);                     // overlap with compute
      __builtin_prefetch(gA + k0 + 64, 0, 3);      // near-cache prefetch
      __builtin_prefetch(gW + k0 + 64, 0, 3);
    }
    v16bf a0 = load_frag_bf16(sA[cur], waveM * 32);
    v16bf a1 = load_frag_bf16(sA[cur], waveM * 32 + 16);
    v16bf b0 = load_frag_bf16(sB[cur], waveN * 32);
    v16bf b1 = load_frag_bf16(sB[cur], waveN * 32 + 16);
    acc[0][0] = __builtin_amdgcn_wmma_f32_16x16x32_bf16(false, a0, false, b0, (short)0, acc[0][0], false, false);
    acc[0][1] = __builtin_amdgcn_wmma_f32_16x16x32_bf16(false, a0, false, b1, (short)0, acc[0][1], false, false);
    acc[1][0] = __builtin_amdgcn_wmma_f32_16x16x32_bf16(false, a1, false, b0, (short)0, acc[1][0], false, false);
    acc[1][1] = __builtin_amdgcn_wmma_f32_16x16x32_bf16(false, a1, false, b1, (short)0, acc[1][1], false, false);
  }
  // C/D layout: VGPR r: lanes0-15 -> M=r, lanes16-31 -> M=8+r; N = lane&15
  const int lane = tid & 31;
  const int cn = lane & 15;
  const int cm = (lane >> 4) << 3;
#pragma unroll
  for (int i = 0; i < 2; ++i)
#pragma unroll
    for (int j = 0; j < 2; ++j) {
      const int rowB = mBase + waveM * 32 + i * 16 + cm;
      const int colB = nBase + waveN * 32 + j * 16 + cn;
#pragma unroll
      for (int r = 0; r < 8; ++r)
        out[(size_t)(rowB + r) * GN + colB] = acc[i][j][r];
    }
}

// ======================= 4) persistent SSM scan ============================
// h_t = h_{t-1} A^T + xb_t, full f32 via v_wmma_f32_16x16x4_f32.
// 16 blocks x 256 threads; wave owns one 16x16 C tile. K split into 4
// independent accumulator chains for WMMA ILP. Device-scope atomic barrier
// per step (16 blocks << #WGPs -> always co-resident, no deadlock).
__global__ __launch_bounds__(256)
void ssm_scan_kernel(const float* __restrict__ An,   // [H,H] f32
                     const float* __restrict__ xb,   // [B*T*H] f32
                     float* __restrict__ hs,         // [B*T*H] f32
                     unsigned* __restrict__ sync) {
  const int tid  = threadIdx.x;
  const int lane = tid & 31;
  const int wid  = tid >> 5;          // 8 waves
  const int mt   = wid >> 2;          // 0..1
  const int nt   = wid & 3;           // 0..3
  const int nBase = blockIdx.x * 64 + nt * 16;
  const int mBase = mt * 16;
  // f32 WMMA 16x4 frag: lane<16 -> K={k0,k0+1}; lane>=16 -> K={k0+2,k0+3}
  const int am   = lane & 15;
  const int koff = (lane >> 4) << 1;
  const int cn   = lane & 15;
  const int cm   = (lane >> 4) << 3;
  const unsigned nb = gridDim.x;

  for (int t = 0; t < T_DIM; ++t) {
    v8f a0 = {}, a1 = {}, a2 = {}, a3 = {};
    if (t > 0) {
      const float* hrow = hs + ((size_t)(mBase + am) * T_DIM + (t - 1)) * H_DIM + koff;
      const float* arow = An + (size_t)(nBase + am) * H_DIM + koff;
      for (int k0 = 0; k0 < H_DIM; k0 += 16) {
        float2 h0 = *(const float2*)(hrow + k0);
        float2 w0 = *(const float2*)(arow + k0);
        float2 h1 = *(const float2*)(hrow + k0 + 4);
        float2 w1 = *(const float2*)(arow + k0 + 4);
        float2 h2 = *(const float2*)(hrow + k0 + 8);
        float2 w2 = *(const float2*)(arow + k0 + 8);
        float2 h3 = *(const float2*)(hrow + k0 + 12);
        float2 w3 = *(const float2*)(arow + k0 + 12);
        v2f av, bv;
        av.x = h0.x; av.y = h0.y; bv.x = w0.x; bv.y = w0.y;
        a0 = __builtin_amdgcn_wmma_f32_16x16x4_f32(false, av, false, bv, (short)0, a0, false, false);
        av.x = h1.x; av.y = h1.y; bv.x = w1.x; bv.y = w1.y;
        a1 = __builtin_amdgcn_wmma_f32_16x16x4_f32(false, av, false, bv, (short)0, a1, false, false);
        av.x = h2.x; av.y = h2.y; bv.x = w2.x; bv.y = w2.y;
        a2 = __builtin_amdgcn_wmma_f32_16x16x4_f32(false, av, false, bv, (short)0, a2, false, false);
        av.x = h3.x; av.y = h3.y; bv.x = w3.x; bv.y = w3.y;
        a3 = __builtin_amdgcn_wmma_f32_16x16x4_f32(false, av, false, bv, (short)0, a3, false, false);
      }
    }
    v8f acc = (a0 + a1) + (a2 + a3);
    // h_t = acc + xb_t ; store into hs
#pragma unroll
    for (int r = 0; r < 8; ++r) {
      const size_t off = ((size_t)(mBase + cm + r) * T_DIM + t) * H_DIM + (nBase + cn);
      hs[off] = acc[r] + xb[off];
    }
    // ---- grid-wide step barrier ----
    __syncthreads();
    if (tid == 0) {
      __threadfence();
      __hip_atomic_fetch_add(&sync[t], 1u, __ATOMIC_RELEASE, __HIP_MEMORY_SCOPE_AGENT);
      while (__hip_atomic_load(&sync[t], __ATOMIC_ACQUIRE, __HIP_MEMORY_SCOPE_AGENT) < nb)
        __builtin_amdgcn_s_sleep(1);
    }
    __syncthreads();
  }
}

// ======================= launcher ==========================================
extern "C" void kernel_launch(void* const* d_in, const int* in_sizes, int n_in,
                              void* d_out, int out_size, void* d_ws, size_t ws_size,
                              hipStream_t stream) {
  (void)in_sizes; (void)n_in; (void)out_size; (void)ws_size;
  const float* x  = (const float*)d_in[0];
  const float* WA = (const float*)d_in[1];
  const float* WB = (const float*)d_in[2];
  const float* WC = (const float*)d_in[3];

  float* ys = (float*)d_out;                                  // [B,T,H]
  float* hs = (float*)d_out + (size_t)B_DIM * T_DIM * H_DIM;  // [B,T,H]

  // workspace (~143 MB): sigmas | sync | A_n f32 | B bf16 | C bf16 | xb f32
  char* ws = (char*)d_ws;
  float*    inv_sigma = (float*)ws;
  unsigned* sync      = (unsigned*)(ws + 64);
  float*    An        = (float*)(ws + 8192);
  __bf16*   Bb        = (__bf16*)(ws + 8192 + (size_t)4 * 1024 * 1024);
  __bf16*   Cb        = (__bf16*)(ws + 8192 + (size_t)6 * 1024 * 1024);
  float*    xb        = (float*)(ws + 8192 + (size_t)8 * 1024 * 1024);

  power_iter_kernel<<<3, 1024, 0, stream>>>(WA, WB, WC, inv_sigma);
  pack_kernel<<<(H_DIM * H_DIM + 255) / 256, 256, 0, stream>>>(
      WA, WB, WC, inv_sigma, An, Bb, Cb, sync);
  gemm_xW_bf16<<<dim3(GM / 128, GN / 128), 512, 0, stream>>>(x, Bb, xb);
  ssm_scan_kernel<<<16, 256, 0, stream>>>(An, xb, hs, sync);
  gemm_xW_bf16<<<dim3(GM / 128, GN / 128), 512, 0, stream>>>(hs, Cb, ys);
}
